// GlobalAttention_19722489823917
// MI455X (gfx1250) — matrix operand order repaired and checked by reading, not compiled, not run on hardware
//
#include <hip/hip_runtime.h>
#include <hip/hip_bf16.h>

typedef __attribute__((ext_vector_type(16))) __bf16 v16bf;
typedef __attribute__((ext_vector_type(8)))  float  v8f;

#define C_IN   64
#define CH     8
#define NH     8
#define S_DIM  256
#define N_DIM  2048
#define INF_F  1e9f
#define EPS_F  1e-10f
#define THREADS 256
#define WAVES   8

// ---- LDS layout (in floats) ----
#define OFF_KV    0                        // [n][0:8]=k, [8:16]=v : 2048*16
#define OFF_SC    (OFF_KV + N_DIM*16)      // scores/probs 8*2048
#define OFF_COLP  (OFF_SC + NH*N_DIM)      // 64 masked colsum (ds_add_f32 target)
#define OFF_QMEAN (OFF_COLP + 64)          // 64
#define OFF_Q     (OFF_QMEAN + 64)         // 64
#define OFF_OP    (OFF_Q + 64)             // 4*64 o partials
#define OFF_O     (OFF_OP + 256)           // 64
#define OFF_MSUM  (OFF_O + 64)             // 1 (+pad)
#define OFF_G     (OFF_MSUM + 64)          // per-wave gated scratch: 8*16*64
#define SMEM_FLOATS (OFF_G + WAVES*16*64)
#define SMEM_BYTES  (SMEM_FLOATS * 4)

// A fragment for v_wmma_f32_16x16x32_bf16: 16x32 chunk, row-major source.
// lane(0..15): row = lane&15, K(e<8)=e+8g, K(e>=8)=e+8+8g with g = lane>>4
__device__ __forceinline__ v16bf load_A_rowmajor(const float* base, int ldr, int k0, int lane) {
  const int g = lane >> 4;
  const int r = lane & 15;
  const float* rp = base + r * ldr + k0 + 8 * g;
  v16bf a;
#pragma unroll
  for (int e = 0; e < 8; e += 2) {
    float2 f = *(const float2*)(rp + e);
    a[e] = (__bf16)f.x; a[e + 1] = (__bf16)f.y;
  }
#pragma unroll
  for (int e = 8; e < 16; e += 2) {
    float2 f = *(const float2*)(rp + e + 8);
    a[e] = (__bf16)f.x; a[e + 1] = (__bf16)f.y;
  }
  return a;
}

// Same, but also fuses the masked column-sum: acc[e] += m[row][K(e)] * mask[row].
// The f32 values are in flight for the bf16 convert anyway -> colsum costs 0 extra bytes.
__device__ __forceinline__ v16bf load_A_accum(const float* base, int ldr, int k0, int lane,
                                              float mw, float* acc) {
  const int g = lane >> 4;
  const int r = lane & 15;
  const float* rp = base + r * ldr + k0 + 8 * g;
  v16bf a;
#pragma unroll
  for (int e = 0; e < 8; e += 2) {
    float2 f = *(const float2*)(rp + e);
    a[e] = (__bf16)f.x; a[e + 1] = (__bf16)f.y;
    acc[e] += f.x * mw;  acc[e + 1] += f.y * mw;
  }
#pragma unroll
  for (int e = 8; e < 16; e += 2) {
    float2 f = *(const float2*)(rp + e + 8);
    a[e] = (__bf16)f.x; a[e + 1] = (__bf16)f.y;
    acc[e] += f.x * mw;  acc[e + 1] += f.y * mw;
  }
  return a;
}

// B fragment: 32x16 chunk of W[64 x ldw] (row-major), columns j0..j0+15.
// lane = col (mod 16), K(e<8) = k0 + e + 8*laneGroup, K(e>=8) = k0 + e + 8 + 8*laneGroup
__device__ __forceinline__ v16bf load_B_coltile(const float* W, int ldw, int k0, int j0, int lane) {
  const int g = lane >> 4;
  const int j = (lane & 15) + j0;
  v16bf b;
#pragma unroll
  for (int e = 0; e < 16; ++e)
    b[e] = (__bf16)W[(k0 + 8 * g + e + (e & 8)) * ldw + j];
  return b;
}

// Fused [Wk | Wv] B fragment: cols 0..7 = Wk (64x8), cols 8..15 = Wv (64x8)
__device__ __forceinline__ v16bf load_B_kv(const float* Wk, const float* Wv, int k0, int lane) {
  const int g = lane >> 4;
  const int j = lane & 15;
  const float* W = (j < 8) ? Wk : Wv;
  const int jj = j & 7;
  v16bf b;
#pragma unroll
  for (int e = 0; e < 16; ++e)
    b[e] = (__bf16)W[(k0 + 8 * g + e + (e & 8)) * CH + jj];
  return b;
}

__device__ __forceinline__ v8f wmma_bf16(v16bf a, v16bf b, v8f c) {
  return __builtin_amdgcn_wmma_f32_16x16x32_bf16(false, a, false, b, (short)0, c, false, false);
}

__global__ void __launch_bounds__(THREADS, 1)
global_attention_kernel(const float* __restrict__ m, const float* __restrict__ mask,
                        const float* __restrict__ Wq, const float* __restrict__ Wk,
                        const float* __restrict__ Wv, const float* __restrict__ Wg,
                        const float* __restrict__ bg, const float* __restrict__ Wo,
                        const float* __restrict__ bo, float* __restrict__ out) {
  extern __shared__ float smem[];
  const int tid  = threadIdx.x;
  const int lane = tid & 31;
  const int wave = tid >> 5;
  const int s    = blockIdx.x;
  const int jcol = lane & 15;
  const int g    = lane >> 4;

  const float* ms    = m    + (size_t)s * N_DIM * C_IN;
  const float* masks = mask + (size_t)s * N_DIM;

  // zero the colsum accumulation target before pass-1 atomics
  if (tid < 64) smem[OFF_COLP + tid] = 0.f;
  __syncthreads();

  // ---------------- Pass 1: k|v = m @ [Wk|Wv] via WMMA, colsum fused ----------------
  v16bf bkv0 = load_B_kv(Wk, Wv, 0, lane);
  v16bf bkv1 = load_B_kv(Wk, Wv, 32, lane);
  float acc0[16], acc1[16];
#pragma unroll
  for (int e = 0; e < 16; ++e) { acc0[e] = 0.f; acc1[e] = 0.f; }

  for (int t = 0; t < 16; ++t) {
    const int row0 = wave * 256 + t * 16;
    const float* mrow = ms + (size_t)row0 * C_IN;
    __builtin_prefetch(mrow + 16 * C_IN, 0, 0);
    const float mw = masks[row0 + (lane & 15)];   // this lane's row mask weight
    v16bf a0 = load_A_accum(mrow, C_IN, 0,  lane, mw, acc0);
    v16bf a1 = load_A_accum(mrow, C_IN, 32, lane, mw, acc1);
    v8f c = {};
    c = wmma_bf16(a0, bkv0, c);
    c = wmma_bf16(a1, bkv1, c);
#pragma unroll
    for (int v = 0; v < 8; ++v)                      // C: lane=col, vgpr=row+8g
      smem[OFF_KV + (size_t)(row0 + v + 8 * g) * 16 + jcol] = c[v];
  }

  // flush per-lane colsum partials (col = (e<8?e:e+8)+8g [+32 for chunk 1]) via ds_add_f32
#pragma unroll
  for (int e = 0; e < 16; ++e) {
    const int col = ((e < 8) ? e : e + 8) + 8 * g;
    atomicAdd(&smem[OFF_COLP + col],      acc0[e]);
    atomicAdd(&smem[OFF_COLP + 32 + col], acc1[e]);
  }

  // mask total (2048 floats, one wave)
  if (tid < 32) {
    float mm = 0.f;
    for (int n = lane; n < N_DIM; n += 32) mm += masks[n];
#pragma unroll
    for (int off = 16; off > 0; off >>= 1) mm += __shfl_xor(mm, off, 32);
    if (lane == 0) smem[OFF_MSUM] = mm;
  }
  __syncthreads();

  if (tid < 64)
    smem[OFF_QMEAN + tid] = smem[OFF_COLP + tid] / (smem[OFF_MSUM] + EPS_F);
  __syncthreads();

  // q = (q_mean @ Wq) * c_h^-0.5  (64 outputs, K=64 — VALU is fine)
  if (tid < 64) {
    float acc = 0.f;
#pragma unroll
    for (int k = 0; k < 64; ++k) acc += smem[OFF_QMEAN + k] * Wq[k * 64 + tid];
    smem[OFF_Q + tid] = acc * 0.35355339059327373f;
  }
  __syncthreads();

  // ---------------- scores[h][n] = q[h]·k[n] + INF*(mask-1) ----------------
  for (int p = tid; p < NH * N_DIM; p += THREADS) {
    const int h = p >> 11, n = p & (N_DIM - 1);
    float acc = INF_F * (masks[n] - 1.0f);
#pragma unroll
    for (int c = 0; c < 8; ++c)
      acc += smem[OFF_Q + h * 8 + c] * smem[OFF_KV + (size_t)n * 16 + c];
    smem[OFF_SC + h * N_DIM + n] = acc;
  }
  __syncthreads();

  // ---------------- softmax over n: one wave per head (wave32 shuffles) ----------------
  {
    const int h = wave;
    float mx = -INF_F;
    for (int n = lane; n < N_DIM; n += 32) mx = fmaxf(mx, smem[OFF_SC + h * N_DIM + n]);
#pragma unroll
    for (int off = 16; off > 0; off >>= 1) mx = fmaxf(mx, __shfl_xor(mx, off, 32));
    float sum = 0.f;
    for (int n = lane; n < N_DIM; n += 32) {
      float e = __expf(smem[OFF_SC + h * N_DIM + n] - mx);
      smem[OFF_SC + h * N_DIM + n] = e;
      sum += e;
    }
#pragma unroll
    for (int off = 16; off > 0; off >>= 1) sum += __shfl_xor(sum, off, 32);
    const float inv = 1.0f / sum;
    for (int n = lane; n < N_DIM; n += 32) smem[OFF_SC + h * N_DIM + n] *= inv;
  }
  __syncthreads();

  // ---------------- o[h][c] = sum_n probs[h][n] * v[n][c] ----------------
  {
    const int p = tid & 63, grp = tid >> 6;
    const int h = p >> 3, c = p & 7;
    float acc = 0.f;
    for (int n = grp; n < N_DIM; n += 4)
      acc += smem[OFF_SC + h * N_DIM + n] * smem[OFF_KV + (size_t)n * 16 + 8 + c];
    smem[OFF_OP + grp * 64 + p] = acc;
  }
  __syncthreads();
  if (tid < 64)
    smem[OFF_O + tid] = smem[OFF_OP + tid] + smem[OFF_OP + 64 + tid] +
                        smem[OFF_OP + 128 + tid] + smem[OFF_OP + 192 + tid];
  __syncthreads();

  // ---------------- Pass 2: g = sigmoid(m@Wg + bg); out = (g*o) @ Wo + bo ----------------
  v16bf bWg[4][2], bWo[4][2];
  float bgv[4], bov[4], omul[4];
#pragma unroll
  for (int jt = 0; jt < 4; ++jt) {
    bWg[jt][0] = load_B_coltile(Wg, 64, 0,  jt * 16, lane);
    bWg[jt][1] = load_B_coltile(Wg, 64, 32, jt * 16, lane);
    bWo[jt][0] = load_B_coltile(Wo, 64, 0,  jt * 16, lane);
    bWo[jt][1] = load_B_coltile(Wo, 64, 32, jt * 16, lane);
    bgv[jt]  = bg[jt * 16 + jcol];
    bov[jt]  = bo[jt * 16 + jcol];
    omul[jt] = smem[OFF_O + jt * 16 + jcol];   // o broadcast over n (col = h*8+c)
  }

  float* gw = smem + OFF_G + wave * (16 * 64);  // per-wave 16x64 gated-tile scratch
  for (int t = 0; t < 16; ++t) {
    const int row0 = wave * 256 + t * 16;
    const float* mrow = ms + (size_t)row0 * C_IN;
    __builtin_prefetch(mrow + 16 * C_IN, 0, 0);
    v16bf a0 = load_A_rowmajor(mrow, C_IN, 0, lane);
    v16bf a1 = load_A_rowmajor(mrow, C_IN, 32, lane);

#pragma unroll
    for (int jt = 0; jt < 4; ++jt) {
      v8f cg = {};
      cg = wmma_bf16(a0, bWg[jt][0], cg);
      cg = wmma_bf16(a1, bWg[jt][1], cg);
#pragma unroll
      for (int v = 0; v < 8; ++v) {
        const float x   = cg[v] + bgv[jt];
        const float sig = 1.0f / (1.0f + __expf(-x));
        gw[(v + 8 * g) * 64 + jt * 16 + jcol] = sig * omul[jt];  // C-layout -> row-major LDS
      }
    }

    // wave-local transpose through LDS (compiler orders via s_wait_dscnt), back as A frags
    v16bf ga0 = load_A_rowmajor(gw, 64, 0, lane);
    v16bf ga1 = load_A_rowmajor(gw, 64, 32, lane);

    float* orow = out + ((size_t)s * N_DIM + row0) * C_IN;
#pragma unroll
    for (int jt = 0; jt < 4; ++jt) {
      v8f cy = {};
      cy = wmma_bf16(ga0, bWo[jt][0], cy);
      cy = wmma_bf16(ga1, bWo[jt][1], cy);
#pragma unroll
      for (int v = 0; v < 8; ++v)
        orow[(size_t)(v + 8 * g) * C_IN + jt * 16 + jcol] = cy[v] + bov[jt];
    }
  }
}

extern "C" void kernel_launch(void* const* d_in, const int* in_sizes, int n_in,
                              void* d_out, int out_size, void* d_ws, size_t ws_size,
                              hipStream_t stream) {
  (void)in_sizes; (void)n_in; (void)out_size; (void)d_ws; (void)ws_size;
  const float* m    = (const float*)d_in[0];
  const float* mask = (const float*)d_in[1];
  const float* Wq   = (const float*)d_in[2];
  const float* Wk   = (const float*)d_in[3];
  const float* Wv   = (const float*)d_in[4];
  const float* Wg   = (const float*)d_in[5];
  const float* bg   = (const float*)d_in[6];
  const float* Wo   = (const float*)d_in[7];
  const float* bo   = (const float*)d_in[8];
  float* out = (float*)d_out;

  // allow >48KB dynamic LDS (CDNA5 WGP has 320KB)
  hipFuncSetAttribute(reinterpret_cast<const void*>(global_attention_kernel),
                      hipFuncAttributeMaxDynamicSharedMemorySize, SMEM_BYTES);

  hipLaunchKernelGGL(global_attention_kernel, dim3(S_DIM), dim3(THREADS), SMEM_BYTES, stream,
                     m, mask, Wq, Wk, Wv, Wg, bg, Wo, bo, out);
}
